// ReactiveChristoffel_2370821948216
// MI455X (gfx1250) — matrix operand order repaired and checked by reading, not compiled, not run on hardware
//
#include <hip/hip_runtime.h>

#define DIMX 1024
#define RANKX 16

typedef _Float16 v8h  __attribute__((ext_vector_type(8)));
typedef _Float16 v16h __attribute__((ext_vector_type(16)));
typedef float    v2f  __attribute__((ext_vector_type(2)));
typedef float    v8f  __attribute__((ext_vector_type(8)));
typedef unsigned int u32x4 __attribute__((ext_vector_type(4)));
typedef unsigned int u32x8 __attribute__((ext_vector_type(8)));

__device__ __forceinline__ float sigmoidf(float z) {
  return 1.0f / (1.0f + __expf(-z));
}

// ---------------------------------------------------------------------------
// Pass 0: one-time f32 -> f16 conversion of gate_w / forget_w into workspace.
// Weights are 8 MB f32 total; f16 copies (4 MB) stay L2-resident and let the
// main kernel pull B tiles with raw async/TDM copies (no VGPR round trip).
// ---------------------------------------------------------------------------
__global__ __launch_bounds__(256)
void rc_cvt_w(const float* __restrict__ gw, const float* __restrict__ fw,
              _Float16* __restrict__ gw16, _Float16* __restrict__ fw16)
{
  const size_t i = ((size_t)blockIdx.x * 256 + threadIdx.x) * 8;  // 8 elems/thread
  const float4* g4 = (const float4*)(gw + i);
  const float4* f4 = (const float4*)(fw + i);
  v8h hg, hf;
#pragma unroll
  for (int q = 0; q < 2; ++q) {
    float4 tg = g4[q], tf = f4[q];
    hg[q*4+0] = (_Float16)tg.x; hg[q*4+1] = (_Float16)tg.y;
    hg[q*4+2] = (_Float16)tg.z; hg[q*4+3] = (_Float16)tg.w;
    hf[q*4+0] = (_Float16)tf.x; hf[q*4+1] = (_Float16)tf.y;
    hf[q*4+2] = (_Float16)tf.z; hf[q*4+3] = (_Float16)tf.w;
  }
  *(v8h*)(gw16 + i) = hg;
  *(v8h*)(fw16 + i) = hf;
}

// ---------------------------------------------------------------------------
// Pass 1: per-token scalars. One wave32 per token.
// ---------------------------------------------------------------------------
__global__ __launch_bounds__(256)
void rc_scalars(const float* __restrict__ v, const float* __restrict__ x,
                const float* __restrict__ U, const float* __restrict__ V_w,
                float* __restrict__ sq_out, float* __restrict__ pm_out,
                float* __restrict__ ts_out, int n_tokens)
{
  const int lane = threadIdx.x & 31;
  const int wave = threadIdx.x >> 5;
  const int tok  = blockIdx.x * 8 + wave;
  if (tok >= n_tokens) return;
  const float* vr = v + (size_t)tok * DIMX;
  const float* xr = x + (size_t)tok * DIMX;

  float proj[RANKX];
#pragma unroll
  for (int r = 0; r < RANKX; ++r) proj[r] = 0.0f;
  float e_acc = 0.0f, p_acc = 0.0f;

  for (int d = lane; d < DIMX; d += 32) {
    float vv = vr[d];
    float xx = xr[d];
    e_acc += vv * vv;
    p_acc += xx * V_w[d];
    const float4* u4 = (const float4*)(U + (size_t)d * RANKX);
#pragma unroll
    for (int q = 0; q < 4; ++q) {
      float4 u = u4[q];
      proj[q * 4 + 0] += vv * u.x;
      proj[q * 4 + 1] += vv * u.y;
      proj[q * 4 + 2] += vv * u.z;
      proj[q * 4 + 3] += vv * u.w;
    }
  }
#pragma unroll
  for (int off = 16; off >= 1; off >>= 1) {
    e_acc += __shfl_xor(e_acc, off, 32);
    p_acc += __shfl_xor(p_acc, off, 32);
#pragma unroll
    for (int r = 0; r < RANKX; ++r) proj[r] += __shfl_xor(proj[r], off, 32);
  }
  if (lane == 0) {
    float nsq = 0.0f;
#pragma unroll
    for (int r = 0; r < RANKX; ++r) nsq += proj[r] * proj[r];
    float scale  = 1.0f / (1.0f + sqrtf(nsq));
    float pot    = sigmoidf(p_acc);
    float energy = tanhf(e_acc * (1.0f / DIMX));
    float sing   = (pot > 0.8f) ? 10.0f : 1.0f;
    pm_out[tok] = 1.0f + pot;
    ts_out[tok] = (1.0f + 0.5f * energy) * sing;
#pragma unroll
    for (int r = 0; r < RANKX; ++r)
      sq_out[(size_t)tok * RANKX + r] = proj[r] * proj[r] * scale;
  }
}

// ---------------------------------------------------------------------------
// Pass 2: fused dual-GEMM (gate+friction) with double-buffered staging.
//   gate  B tile : TDM tensor_load_to_lds (wave 0, TENSORcnt), HW LDS padding
//   forget B tile: per-lane global_load_async_to_lds_b128 (ASYNCcnt)
//   x (A) tile   : global f32 load -> cvt f16 -> ds_store (double-buffered)
// Workgroup tile: 128 tokens x 64 cols. Wave w owns rows [16w,16w+16).
// ---------------------------------------------------------------------------
#define A_STRIDE 40   // f16/row in LDS (32 data + 4-dword pad) = 80B, bank-clean

__global__ __launch_bounds__(256)
void rc_main(const float* __restrict__ v,  const float* __restrict__ x,
             const float* __restrict__ W,
             const _Float16* __restrict__ gw16, const float* __restrict__ gate_b,
             const _Float16* __restrict__ fw16, const float* __restrict__ forget_b,
             const float* __restrict__ sq_ws, const float* __restrict__ pm_ws,
             const float* __restrict__ ts_ws, float* __restrict__ out)
{
  __shared__ _Float16 a_lds[2][128 * A_STRIDE];
  __shared__ _Float16 bg_lds[2][64 * A_STRIDE];
  __shared__ _Float16 bf_lds[2][64 * A_STRIDE];
  __shared__ float sq_lds[128 * RANKX];
  __shared__ float wt_lds[64 * RANKX];
  __shared__ float pm_lds[128];
  __shared__ float ts_lds[128];

  const int tid  = threadIdx.x;
  const int lane = tid & 31;
  const int wave = tid >> 5;
  const int N0   = blockIdx.x * 64;
  const int M0   = blockIdx.y * 128;

  // ---- prologue: stage per-token scalars and W rows for our 64 columns ----
  for (int i = tid; i < 128 * RANKX; i += 256) sq_lds[i] = sq_ws[(size_t)M0 * RANKX + i];
  for (int i = tid; i < 64 * RANKX;  i += 256) wt_lds[i] = W[(size_t)N0 * RANKX + i];
  for (int i = tid; i < 128;         i += 256) { pm_lds[i] = pm_ws[M0 + i]; ts_lds[i] = ts_ws[M0 + i]; }

  v8f gate_acc[4], fric_acc[4], gam_acc[4];
#pragma unroll
  for (int s = 0; s < 4; ++s) {
    gate_acc[s] = (v8f)0.0f; fric_acc[s] = (v8f)0.0f; gam_acc[s] = (v8f)0.0f;
  }

  // ---- staging coordinates ----
  const int arow  = tid >> 1;            // 0..127
  const int ahalf = (tid & 1) * 16;      // 0 / 16
  const float*  a_src = x + (size_t)(M0 + arow) * DIMX + ahalf;
  const int brow = tid >> 2;             // 0..63
  const int bseg = (tid & 3) * 8;        // 0,8,16,24

  // forget-matrix async-to-LDS: per-lane b128 (8 f16) into padded LDS rows
  const unsigned f_voff_base = (unsigned)(((N0 + brow) * DIMX + bseg) * 2);
  const unsigned f_lds_off[2] = {
    (unsigned)(unsigned long long)(&bf_lds[0][brow * A_STRIDE + bseg]),
    (unsigned)(unsigned long long)(&bf_lds[1][brow * A_STRIDE + bseg])
  };

  // gate-matrix TDM descriptor (uniform; group1 is loop-invariant)
  const unsigned long long g_tile_base =
      (unsigned long long)gw16 + (unsigned long long)(N0) * DIMX * 2;
  const unsigned g_lds_off[2] = {
    (unsigned)(unsigned long long)(&bg_lds[0][0]),
    (unsigned)(unsigned long long)(&bg_lds[1][0])
  };
  u32x8 tdm_g1;
  tdm_g1[0] = (1u << 16)        // data_size = 2 bytes
            | (1u << 20)        // pad_enable
            | (3u << 22)        // pad_interval: 16 dwords (64B = one tile row)
            | (3u << 25);       // pad_amount: 4 dwords (16B) -> 80B LDS row stride
  tdm_g1[1] = (1024u & 0xFFFFu) << 16;          // tensor_dim0[15:0] = 1024
  tdm_g1[2] = (1024u & 0xFFFFu) << 16;          // tensor_dim1[15:0] = 1024
  tdm_g1[3] = (32u << 16);                      // tile_dim0 = 32 elems (64B row)
  tdm_g1[4] = 64u;                              // tile_dim1 = 64 rows
  tdm_g1[5] = 1024u;                            // tensor_dim0_stride = 1024 elems
  tdm_g1[6] = 0u;
  tdm_g1[7] = 0u;

  auto issue_B = [&](int kk, int buf) {
    // forget: per-lane async copy (ASYNCcnt)
    unsigned voff = f_voff_base + (unsigned)kk * 2u;
    asm volatile("global_load_async_to_lds_b128 %0, %1, %2"
                 :: "v"(f_lds_off[buf]), "v"(voff), "s"(fw16) : "memory");
    // gate: one TDM transfer for the whole 64x32 f16 tile (TENSORcnt)
    if (wave == 0) {
      unsigned long long ga = g_tile_base + (unsigned long long)kk * 2u;
      u32x4 g0;
      g0[0] = 1u;                                        // count=1, user mode
      g0[1] = g_lds_off[buf];                            // lds_addr
      g0[2] = (unsigned)(ga & 0xFFFFFFFFu);              // global_addr[31:0]
      g0[3] = (unsigned)((ga >> 32) & 0x1FFFFFFu)        // global_addr[56:32]
            | 0x80000000u;                               // type=2 ("image")
      asm volatile("tensor_load_to_lds %0, %1" :: "s"(g0), "s"(tdm_g1) : "memory");
    }
  };

  float4 a_pref[4];
  auto load_A = [&](int kk) {
    const float4* a4 = (const float4*)(a_src + kk);
#pragma unroll
    for (int q = 0; q < 4; ++q) a_pref[q] = a4[q];
  };
  auto store_A = [&](int buf) {
    v8h h0, h1;
#pragma unroll
    for (int q = 0; q < 2; ++q) {
      float4 t = a_pref[q];
      h0[q*4+0] = (_Float16)t.x; h0[q*4+1] = (_Float16)t.y;
      h0[q*4+2] = (_Float16)t.z; h0[q*4+3] = (_Float16)t.w;
    }
#pragma unroll
    for (int q = 0; q < 2; ++q) {
      float4 t = a_pref[2 + q];
      h1[q*4+0] = (_Float16)t.x; h1[q*4+1] = (_Float16)t.y;
      h1[q*4+2] = (_Float16)t.z; h1[q*4+3] = (_Float16)t.w;
    }
    _Float16* ad = &a_lds[buf][arow * A_STRIDE + ahalf];
    *(v8h*)(ad)     = h0;
    *(v8h*)(ad + 8) = h1;
  };

  // ---- software pipeline prologue: fill buffer 0 ----
  issue_B(0, 0);
  load_A(0);
  store_A(0);
  asm volatile("s_wait_asynccnt 0" ::: "memory");
  if (wave == 0) __builtin_amdgcn_s_wait_tensorcnt(0);
  __syncthreads();

  // ---- K loop: 1024 in steps of 32, double-buffered ----
  for (int kk = 0; kk < DIMX; kk += 32) {
    const int cur = (kk >> 5) & 1;
    const int nxt = cur ^ 1;
    const bool has_next = (kk + 32) < DIMX;
    if (has_next) {
      issue_B(kk + 32, nxt);    // ASYNCcnt + TENSORcnt in flight behind WMMAs
      load_A(kk + 32);          // LOADcnt in flight behind WMMAs
    }

    // A fragment (f16 16x32): lane L -> M=L%16, 16 contiguous K at (L>>4)*16
    union { v16h v; v8h h[2]; } au;
    const int abase = (wave * 16 + (lane & 15)) * A_STRIDE + ((lane >> 4) * 16);
    au.h[0] = *(const v8h*)&a_lds[cur][abase];
    au.h[1] = *(const v8h*)&a_lds[cur][abase + 8];

#pragma unroll
    for (int sub = 0; sub < 4; ++sub) {
      // B fragment (f16 32x16): lane L -> N=L%16, 16 contiguous K at (L>>4)*16
      const int bbase = (sub * 16 + (lane & 15)) * A_STRIDE + ((lane >> 4) * 16);
      union { v16h v; v8h h[2]; } bu, fu;
      bu.h[0] = *(const v8h*)&bg_lds[cur][bbase];
      bu.h[1] = *(const v8h*)&bg_lds[cur][bbase + 8];
      fu.h[0] = *(const v8h*)&bf_lds[cur][bbase];
      fu.h[1] = *(const v8h*)&bf_lds[cur][bbase + 8];
      gate_acc[sub] = __builtin_amdgcn_wmma_f32_16x16x32_f16(
          false, au.v, false, bu.v, (short)0, gate_acc[sub], false, false);
      fric_acc[sub] = __builtin_amdgcn_wmma_f32_16x16x32_f16(
          false, au.v, false, fu.v, (short)0, fric_acc[sub], false, false);
    }

    if (has_next) {
      store_A(nxt);
      asm volatile("s_wait_asynccnt 0" ::: "memory");
      if (wave == 0) __builtin_amdgcn_s_wait_tensorcnt(0);
    }
    __syncthreads();
  }

  // ---- gamma = sq[128x16] @ W^T[16x64] via v_wmma_f32_16x16x4_f32 ----
  v2f asq[4];
#pragma unroll
  for (int kc = 0; kc < 4; ++kc)
    asq[kc] = *(const v2f*)&sq_lds[(wave * 16 + (lane & 15)) * RANKX + kc * 4 + (lane >> 4) * 2];
#pragma unroll
  for (int sub = 0; sub < 4; ++sub) {
#pragma unroll
    for (int kc = 0; kc < 4; ++kc) {
      v2f bw = *(const v2f*)&wt_lds[(sub * 16 + (lane & 15)) * RANKX + kc * 4 + (lane >> 4) * 2];
      gam_acc[sub] = __builtin_amdgcn_wmma_f32_16x16x4_f32(
          false, asq[kc], false, bw, (short)0, gam_acc[sub], false, false);
    }
  }

  // ---- fused epilogue ----
  // f32 C/D layout: VGPR i, lane L -> M = i + 8*(L>=16), N = L%16
#pragma unroll
  for (int sub = 0; sub < 4; ++sub) {
    const int col = N0 + sub * 16 + (lane & 15);
    const float gb = gate_b[col];
    const float fb = forget_b[col];
#pragma unroll
    for (int i = 0; i < 8; ++i) {
      const int rl = wave * 16 + i + ((lane >> 4) << 3);
      const int n  = M0 + rl;
      float gate = sigmoidf(gate_acc[sub][i] + gb);
      float fric = sigmoidf(fric_acc[sub][i] + fb);
      float gm   = gam_acc[sub][i] * pm_lds[rl];
      gm = fminf(fmaxf(gm, -5.0f), 5.0f);
      float o = (gm * gate + fric * v[(size_t)n * DIMX + col]) * ts_lds[rl];
      out[(size_t)n * DIMX + col] = o;
    }
  }
}

// ---------------------------------------------------------------------------
extern "C" void kernel_launch(void* const* d_in, const int* in_sizes, int n_in,
                              void* d_out, int out_size, void* d_ws, size_t ws_size,
                              hipStream_t stream)
{
  const float* v  = (const float*)d_in[0];
  const float* x  = (const float*)d_in[1];
  const float* U  = (const float*)d_in[2];
  const float* W  = (const float*)d_in[3];
  const float* Vw = (const float*)d_in[4];
  const float* gw = (const float*)d_in[5];
  const float* gb = (const float*)d_in[6];
  const float* fw = (const float*)d_in[7];
  const float* fb = (const float*)d_in[8];
  float* out = (float*)d_out;

  const int n_tokens = in_sizes[0] / DIMX;   // 32768 for B=8,S=4096

  // workspace: [gw16 2MB][fw16 2MB][sq 2MB][pm 128KB][ts 128KB]
  char* ws = (char*)d_ws;
  _Float16* gw16 = (_Float16*)ws;
  _Float16* fw16 = gw16 + (size_t)DIMX * DIMX;
  float* sq = (float*)(ws + 2 * (size_t)DIMX * DIMX * sizeof(_Float16));
  float* pm = sq + (size_t)n_tokens * RANKX;
  float* ts = pm + n_tokens;

  rc_cvt_w<<<(DIMX * DIMX) / (256 * 8), 256, 0, stream>>>(gw, fw, gw16, fw16);
  rc_scalars<<<(n_tokens + 7) / 8, 256, 0, stream>>>(v, x, U, Vw, sq, pm, ts, n_tokens);

  dim3 grid(DIMX / 64, n_tokens / 128);      // 16 x 256 workgroups
  rc_main<<<grid, 256, 0, stream>>>(v, x, W, gw16, gb, fw16, fb, sq, pm, ts, out);
}
